// CrossAttention_1297080123769
// MI455X (gfx1250) — matrix operand order repaired and checked by reading, not compiled, not run on hardware
//
#include <hip/hip_runtime.h>
#include <hip/hip_bf16.h>

// ---------------------------------------------------------------------------
// CDNA5 (gfx1250, wave32) cross-attention:
//   q = (x Wq^T)*scale, k = ctx Wk^T, v = ctx Wv^T     (WMMA f16->f32 GEMMs)
//   attn = softmax(q k^T) v                            (flash-style, WMMA)
//   out = attn Wout^T + b                              (WMMA GEMM, f32 out)
// Matrix math: v_wmma_f32_16x16x32_f16.
// Shared tiles staged via global_load_async_to_lds_b128 (ASYNCcnt) with
// double buffering; fragments then come from ds_load_b128.
// ---------------------------------------------------------------------------

typedef _Float16 h4   __attribute__((ext_vector_type(4)));
typedef _Float16 h8   __attribute__((ext_vector_type(8)));
typedef _Float16 v16h __attribute__((ext_vector_type(16)));
typedef float    v8f  __attribute__((ext_vector_type(8)));

#define DIM_   512
#define LSEQ_  4096          // S == C == 4096
#define BATCH_ 2
#define HEADS_ 8
#define DH_    64
#define SCALE_ 0.125f        // 64^-0.5

union frag_u { v16h v; h8 h[2]; };

// A-fragment (16x32 f16, M x K): lane&15 = row, chunks at k = half*8 and +16.
__device__ __forceinline__ v16h load_a16(const _Float16* a, int lda, int lane) {
  const int row = lane & 15, c0 = (lane >> 4) << 3;
  frag_u u;
  u.h[0] = *(const h8*)(a + (size_t)row * lda + c0);
  u.h[1] = *(const h8*)(a + (size_t)row * lda + c0 + 16);
  return u.v;
}

// B-fragment (32x16 f16, K x N) from B^T (row-major [N][K]):
// lane&15 = column n, chunks at k = half*8 and +16.
__device__ __forceinline__ v16h load_b16(const _Float16* bt, int ldb, int lane) {
  const int col = lane & 15, k0 = (lane >> 4) << 3;
  const h8* p = (const h8*)(bt + (size_t)col * ldb + k0);
  frag_u u;
  u.h[0] = p[0];
  u.h[1] = p[2];
  return u.v;
}

__device__ __forceinline__ v8f wmma16(v16h a, v16h b, v8f c) {
  return __builtin_amdgcn_wmma_f32_16x16x32_f16(false, a, false, b,
                                                (short)0, c, false, false);
}

// gfx1250 async copy: 16B/lane global -> LDS, tracked by ASYNCcnt.
__device__ __forceinline__ void async_ld16(uint32_t lds_off, const _Float16* g) {
  asm volatile("global_load_async_to_lds_b128 %0, %1, off"
               :: "v"(lds_off), "v"((uint64_t)(uintptr_t)g) : "memory");
}
#define WAIT_ASYNC(n) asm volatile("s_wait_asynccnt " #n ::: "memory")

// ---------------------------------------------------------------------------
__global__ void cvt_f32_to_f16(const float* __restrict__ s,
                               _Float16* __restrict__ d, int n4) {
  int i = blockIdx.x * blockDim.x + threadIdx.x;
  if (i >= n4) return;
  float4 v = ((const float4*)s)[i];
  h4 o = { (_Float16)v.x, (_Float16)v.y, (_Float16)v.z, (_Float16)v.w };
  ((h4*)d)[i] = o;
}

// ---------------------------------------------------------------------------
// C[M=8192 x 512] = A * W^T, W stored [512 out][512 in] f16.
// B-tile (64 rows x 32 k) staged in LDS via async copies, double buffered.
// transposeV==0 -> store [b,h,l,64]; ==1 -> store [b,h,64,l] (V^T).
__global__ void gemm_qkv(const _Float16* __restrict__ A,
                         const _Float16* __restrict__ W,
                         _Float16* __restrict__ outp,
                         int transposeV, float scale) {
  __shared__ __align__(16) _Float16 bs[2][64 * 32];   // 4KB per buffer
  const int tid = threadIdx.x;
  const int wave = tid >> 5, lane = tid & 31;
  const int rm0 = blockIdx.x * 128 + wave * 16;
  const int n0  = blockIdx.y * 64;

  // staging: 256 threads x 16B = one 64x32 f16 tile
  const int srow = tid >> 2, schunk = tid & 3;
  const _Float16* gsrc = W + (size_t)(n0 + srow) * DIM_ + schunk * 8;
  const uint32_t lofs = (uint32_t)(srow * 64 + schunk * 16);   // bytes in tile
  const uint32_t lb[2] = { (uint32_t)(uintptr_t)&bs[0][0] + lofs,
                           (uint32_t)(uintptr_t)&bs[1][0] + lofs };

  async_ld16(lb[0], gsrc);                 // stage kk = 0
  v8f acc[4] = {};
  const _Float16* arow = A + (size_t)rm0 * DIM_;
  int cur = 0;
  for (int kk = 0; kk < DIM_; kk += 32) {
    if (kk + 32 < DIM_) {
      async_ld16(lb[cur ^ 1], gsrc + kk + 32);
      WAIT_ASYNC(1);                       // current tile landed, next in flight
    } else {
      WAIT_ASYNC(0);
    }
    __syncthreads();
    v16h a = load_a16(arow + kk, DIM_, lane);
    const _Float16* bt = &bs[cur][0];
    v16h b[4];
#pragma unroll
    for (int j = 0; j < 4; ++j) b[j] = load_b16(bt + j * 16 * 32, 32, lane);
#pragma unroll
    for (int j = 0; j < 4; ++j) acc[j] = wmma16(a, b[j], acc[j]);
    __syncthreads();                       // everyone done reading before rewrite
    cur ^= 1;
  }

  const int hf = lane >> 4, lm = lane & 15;
  const int bb = rm0 >> 12;                // / 4096
  const int l0 = rm0 & (LSEQ_ - 1);
#pragma unroll
  for (int j = 0; j < 4; ++j) {
    const int n = n0 + j * 16 + lm;
    const int h = n >> 6, dh = n & 63;
#pragma unroll
    for (int r = 0; r < 8; ++r) {
      const int l = l0 + hf * 8 + r;
      const float v = acc[j][r] * scale;
      if (transposeV)
        outp[((size_t)(bb * HEADS_ + h) * DH_ + dh) * LSEQ_ + l] = (_Float16)v;
      else
        outp[((size_t)(bb * HEADS_ + h) * LSEQ_ + l) * DH_ + dh] = (_Float16)v;
    }
  }
}

// ---------------------------------------------------------------------------
// Flash attention: grid (S/64, B*H), block 128 (4 waves x 16 query rows).
// K-tile and V^T-tile (64x64 f16 each) staged via async copies, shared by
// all 4 waves, double buffered.
__global__ void flash_attn(const _Float16* __restrict__ Q,
                           const _Float16* __restrict__ K,
                           const _Float16* __restrict__ Vt,
                           _Float16* __restrict__ O) {
  __shared__ __align__(16) _Float16 kb[2][64 * 64];   // 8KB per buffer
  __shared__ __align__(16) _Float16 vb[2][64 * 64];
  __shared__ __align__(16) _Float16 pbuf[4][16 * 64]; // per-wave P relayout
  const int tid = threadIdx.x;
  const int wave = tid >> 5, lane = tid & 31;
  const int hf = lane >> 4, lm = lane & 15;
  const int bh = blockIdx.y;
  const int b = bh >> 3, h = bh & 7;
  const int s0 = blockIdx.x * 64 + wave * 16;

  const _Float16* qbase = Q + ((size_t)bh * LSEQ_ + s0) * DH_;
  v16h aq[2];
#pragma unroll
  for (int t = 0; t < 2; ++t) aq[t] = load_a16(qbase + t * 32, DH_, lane);

  float mrow[8], lrow[8];
#pragma unroll
  for (int r = 0; r < 8; ++r) { mrow[r] = -1e30f; lrow[r] = 0.f; }
  v8f o[4] = {};

  // staging bases: tile = 64 rows x 128B; 512 chunks of 16B; 128 thr x 4
  const _Float16* kg = K  + (size_t)bh * LSEQ_ * DH_;   // row c, stride 64
  const _Float16* vg = Vt + (size_t)bh * DH_ * LSEQ_;   // row d, stride 4096
  const uint32_t kbb[2] = { (uint32_t)(uintptr_t)&kb[0][0],
                            (uint32_t)(uintptr_t)&kb[1][0] };
  const uint32_t vbb[2] = { (uint32_t)(uintptr_t)&vb[0][0],
                            (uint32_t)(uintptr_t)&vb[1][0] };
  auto stage = [&](int buf, int cc) {
#pragma unroll
    for (int c = 0; c < 4; ++c) {
      const int q = tid + c * 128;
      const int row = q >> 3, off = (q & 7) * 8;        // off in halves
      const uint32_t lo = (uint32_t)(row * 128 + (q & 7) * 16);
      async_ld16(kbb[buf] + lo, kg + (size_t)(cc + row) * DH_ + off);
      async_ld16(vbb[buf] + lo, vg + (size_t)row * LSEQ_ + cc + off);
    }
  };

  stage(0, 0);
  _Float16* pb = &pbuf[wave][0];
  int cur = 0;
  for (int c0 = 0; c0 < LSEQ_; c0 += 64) {
    if (c0 + 64 < LSEQ_) {
      stage(cur ^ 1, c0 + 64);
      WAIT_ASYNC(8);                      // 8 ops/wave per stage
    } else {
      WAIT_ASYNC(0);
    }
    __syncthreads();
    const _Float16* kt = &kb[cur][0];
    const _Float16* vt = &vb[cur][0];

    // s = q k^T  (16 x 64 f32)
    v8f s[4] = {};
#pragma unroll
    for (int t = 0; t < 2; ++t) {
      v16h bk[4];
#pragma unroll
      for (int j = 0; j < 4; ++j)
        bk[j] = load_b16(kt + j * 16 * 64 + t * 32, 64, lane);
#pragma unroll
      for (int j = 0; j < 4; ++j) s[j] = wmma16(aq[t], bk[j], s[j]);
    }

    // online softmax: row r in VGPR r, cols over 16 lanes x 4 accs
#pragma unroll
    for (int r = 0; r < 8; ++r) {
      float mx = fmaxf(fmaxf(s[0][r], s[1][r]), fmaxf(s[2][r], s[3][r]));
#pragma unroll
      for (int d = 1; d < 16; d <<= 1) mx = fmaxf(mx, __shfl_xor(mx, d, 32));
      const float mnew = fmaxf(mrow[r], mx);
      const float corr = __expf(mrow[r] - mnew);
      float rs = 0.f;
#pragma unroll
      for (int j = 0; j < 4; ++j) {
        float p = __expf(s[j][r] - mnew);
        s[j][r] = p;
        rs += p;
      }
#pragma unroll
      for (int d = 1; d < 16; d <<= 1) rs += __shfl_xor(rs, d, 32);
      lrow[r] = lrow[r] * corr + rs;
      mrow[r] = mnew;
#pragma unroll
      for (int j = 0; j < 4; ++j) o[j][r] *= corr;
    }

    // relayout P (C-fragment -> A-fragment) via wave-private LDS slice
#pragma unroll
    for (int j = 0; j < 4; ++j)
#pragma unroll
      for (int r = 0; r < 8; ++r)
        pb[(hf * 8 + r) * 64 + j * 16 + lm] = (_Float16)s[j][r];

    asm volatile("s_wait_dscnt 0" ::: "memory");   // wave-private RAW via LDS

    v16h ap[2];
#pragma unroll
    for (int t = 0; t < 2; ++t) ap[t] = load_a16(pb + t * 32, 64, lane);

    // o += P V
#pragma unroll
    for (int t = 0; t < 2; ++t) {
      v16h bv[4];
#pragma unroll
      for (int j = 0; j < 4; ++j)
        bv[j] = load_b16(vt + j * 16 * 64 + t * 32, 64, lane);
#pragma unroll
      for (int j = 0; j < 4; ++j) o[j] = wmma16(ap[t], bv[j], o[j]);
    }
    __syncthreads();
    cur ^= 1;
  }

  // normalize and emit head-concat layout [b, s, h*64+d] in f16
#pragma unroll
  for (int r = 0; r < 8; ++r) {
    const float inv = 1.0f / lrow[r];
    const int srow = s0 + hf * 8 + r;
#pragma unroll
    for (int j = 0; j < 4; ++j) {
      const int col = h * DH_ + j * 16 + lm;
      O[((size_t)b * LSEQ_ + srow) * (HEADS_ * DH_) + col] =
          (_Float16)(o[j][r] * inv);
    }
  }
}

// ---------------------------------------------------------------------------
// out[8192 x 512] f32 = A[8192 x 512] f16 * Wout^T + bias
__global__ void gemm_out(const _Float16* __restrict__ A,
                         const _Float16* __restrict__ W,
                         const float* __restrict__ bias,
                         float* __restrict__ outp) {
  __shared__ __align__(16) _Float16 bs[2][64 * 32];
  const int tid = threadIdx.x;
  const int wave = tid >> 5, lane = tid & 31;
  const int rm0 = blockIdx.x * 128 + wave * 16;
  const int n0  = blockIdx.y * 64;

  const int srow = tid >> 2, schunk = tid & 3;
  const _Float16* gsrc = W + (size_t)(n0 + srow) * DIM_ + schunk * 8;
  const uint32_t lofs = (uint32_t)(srow * 64 + schunk * 16);
  const uint32_t lb[2] = { (uint32_t)(uintptr_t)&bs[0][0] + lofs,
                           (uint32_t)(uintptr_t)&bs[1][0] + lofs };

  async_ld16(lb[0], gsrc);
  v8f acc[4] = {};
  const _Float16* arow = A + (size_t)rm0 * DIM_;
  int cur = 0;
  for (int kk = 0; kk < DIM_; kk += 32) {
    if (kk + 32 < DIM_) {
      async_ld16(lb[cur ^ 1], gsrc + kk + 32);
      WAIT_ASYNC(1);
    } else {
      WAIT_ASYNC(0);
    }
    __syncthreads();
    v16h a = load_a16(arow + kk, DIM_, lane);
    const _Float16* bt = &bs[cur][0];
    v16h b[4];
#pragma unroll
    for (int j = 0; j < 4; ++j) b[j] = load_b16(bt + j * 16 * 32, 32, lane);
#pragma unroll
    for (int j = 0; j < 4; ++j) acc[j] = wmma16(a, b[j], acc[j]);
    __syncthreads();
    cur ^= 1;
  }

  const int hf = lane >> 4, lm = lane & 15;
#pragma unroll
  for (int j = 0; j < 4; ++j) {
    const int n = n0 + j * 16 + lm;
    const float bv = bias[n];
#pragma unroll
    for (int r = 0; r < 8; ++r) {
      const int g = rm0 + hf * 8 + r;
      outp[(size_t)g * DIM_ + n] = acc[j][r] + bv;
    }
  }
}

// ---------------------------------------------------------------------------
extern "C" void kernel_launch(void* const* d_in, const int* in_sizes, int n_in,
                              void* d_out, int out_size, void* d_ws,
                              size_t ws_size, hipStream_t stream) {
  (void)in_sizes; (void)n_in; (void)out_size; (void)ws_size;
  const float* x   = (const float*)d_in[0];
  const float* ctx = (const float*)d_in[1];
  const float* wq  = (const float*)d_in[2];
  const float* wk  = (const float*)d_in[3];
  const float* wv  = (const float*)d_in[4];
  const float* wo  = (const float*)d_in[5];
  const float* bo  = (const float*)d_in[6];
  float* out = (float*)d_out;

  char* p = (char*)d_ws;
  auto take = [&](size_t elems) {
    _Float16* r = (_Float16*)p;
    p += (elems * sizeof(_Float16) + 255) & ~(size_t)255;
    return r;
  };
  const size_t NX = (size_t)BATCH_ * LSEQ_ * DIM_;
  const size_t NW = (size_t)DIM_ * DIM_;
  const size_t NH = (size_t)BATCH_ * HEADS_ * LSEQ_ * DH_;

  _Float16* xh  = take(NX);
  _Float16* ch  = take(NX);
  _Float16* wqh = take(NW);
  _Float16* wkh = take(NW);
  _Float16* wvh = take(NW);
  _Float16* woh = take(NW);
  _Float16* Qh  = take(NH);
  _Float16* Kh  = take(NH);
  _Float16* Vth = take(NH);
  _Float16* Oh  = take(NX);

  const dim3 cb(256);
  cvt_f32_to_f16<<<dim3((unsigned)((NX / 4 + 255) / 256)), cb, 0, stream>>>(x, xh, (int)(NX / 4));
  cvt_f32_to_f16<<<dim3((unsigned)((NX / 4 + 255) / 256)), cb, 0, stream>>>(ctx, ch, (int)(NX / 4));
  cvt_f32_to_f16<<<dim3((unsigned)((NW / 4 + 255) / 256)), cb, 0, stream>>>(wq, wqh, (int)(NW / 4));
  cvt_f32_to_f16<<<dim3((unsigned)((NW / 4 + 255) / 256)), cb, 0, stream>>>(wk, wkh, (int)(NW / 4));
  cvt_f32_to_f16<<<dim3((unsigned)((NW / 4 + 255) / 256)), cb, 0, stream>>>(wv, wvh, (int)(NW / 4));
  cvt_f32_to_f16<<<dim3((unsigned)((NW / 4 + 255) / 256)), cb, 0, stream>>>(wo, woh, (int)(NW / 4));

  const dim3 gp(BATCH_ * LSEQ_ / 128, DIM_ / 64);       // (64, 8), 8 waves/blk
  gemm_qkv<<<gp, 256, 0, stream>>>(xh, wqh, Qh, 0, SCALE_);
  gemm_qkv<<<gp, 256, 0, stream>>>(ch, wkh, Kh, 0, 1.0f);
  gemm_qkv<<<gp, 256, 0, stream>>>(ch, wvh, Vth, 1, 1.0f);

  flash_attn<<<dim3(LSEQ_ / 64, BATCH_ * HEADS_), 128, 0, stream>>>(Qh, Kh, Vth, Oh);

  gemm_out<<<gp, 256, 0, stream>>>(Oh, woh, bo, out);
}
